// PerceiverRotary_34239479284271
// MI455X (gfx1250) — matrix operand order repaired and checked by reading, not compiled, not run on hardware
//
#include <hip/hip_runtime.h>

typedef __attribute__((ext_vector_type(16))) __bf16 v16bf;
typedef __attribute__((ext_vector_type(8)))  float  v8f;

#define USE_ASYNC_COPY 1

// ---------------------------------------------------------------------------
// helpers
// ---------------------------------------------------------------------------
__device__ __forceinline__ unsigned short f2bf(float f) {
  unsigned int u = __float_as_uint(f);
  u += 0x7FFFu + ((u >> 16) & 1u);            // round-to-nearest-even
  return (unsigned short)(u >> 16);
}

#if USE_ASYNC_COPY
// CDNA5 async global->LDS copy (ASYNCcnt); per-lane 16B gather/scatter.
__device__ __forceinline__ void async_b128(void* lds_dst, const void* gsrc) {
  unsigned loff = (unsigned)(unsigned long long)lds_dst;
  asm volatile("global_load_async_to_lds_b128 %0, %1, off"
               :: "v"(loff), "v"(gsrc)
               : "memory");
}
__device__ __forceinline__ void wait_async0() {
#if __has_builtin(__builtin_amdgcn_s_wait_asynccnt)
  __builtin_amdgcn_s_wait_asynccnt(0);
#else
  asm volatile("s_wait_asynccnt 0x0" ::: "memory");
#endif
}
#endif

// ---------------------------------------------------------------------------
// elementwise kernels
// ---------------------------------------------------------------------------
__global__ void k_copy_f32(const float* __restrict__ x, float* __restrict__ y, int n) {
  int i = blockIdx.x * blockDim.x + threadIdx.x;
  if (i < n) y[i] = x[i];
}

// f32 (R x C) -> bf16 transposed (C x R)
__global__ void k_cvt_t(const float* __restrict__ x, unsigned short* __restrict__ y,
                        int R, int C) {
  int i = blockIdx.x * blockDim.x + threadIdx.x;
  if (i >= R * C) return;
  int r = i / C, c = i - r * C;
  y[(size_t)c * R + r] = f2bf(x[i]);
}

// LayerNorm over rows of 512, one wave32 per row, bf16 output.
__global__ void k_layernorm_bf16(const float* __restrict__ x,
                                 const float* __restrict__ g,
                                 const float* __restrict__ b,
                                 unsigned short* __restrict__ y, int rows) {
  int wave = (int)((blockIdx.x * blockDim.x + threadIdx.x) >> 5);
  int lane = threadIdx.x & 31;
  if (wave >= rows) return;
  const float* xr = x + (size_t)wave * 512;
  float v[16];
  float s = 0.f;
#pragma unroll
  for (int i = 0; i < 16; ++i) { v[i] = xr[lane + 32 * i]; s += v[i]; }
#pragma unroll
  for (int m = 16; m >= 1; m >>= 1) s += __shfl_xor(s, m, 32);
  float mean = s * (1.f / 512.f);
  float var = 0.f;
#pragma unroll
  for (int i = 0; i < 16; ++i) { float d = v[i] - mean; var += d * d; }
#pragma unroll
  for (int m = 16; m >= 1; m >>= 1) var += __shfl_xor(var, m, 32);
  float rstd = rsqrtf(var * (1.f / 512.f) + 1e-5f);
  unsigned short* yr = y + (size_t)wave * 512;
#pragma unroll
  for (int i = 0; i < 16; ++i) {
    int c = lane + 32 * i;
    yr[c] = f2bf((v[i] - mean) * rstd * g[c] + b[c]);
  }
}

// RoPE (interleaved pairs) + bf16 convert; do_rope==0 -> convert only.
// x: rows x cols with row stride ld_in.
// trans==0: y[r*cols + c]  (row-major)
// trans==1: y[((b*cols)+c)*rpb + n]  where r = b*rpb + n  (per-batch transpose)
__global__ void k_rope_bf16(const float* __restrict__ x, int ld_in,
                            unsigned short* __restrict__ y,
                            const float* __restrict__ ts,
                            const float* __restrict__ inv_freq,
                            float sign, int rows, int cols, int do_rope,
                            int trans, int rpb) {
  int i = blockIdx.x * blockDim.x + threadIdx.x;
  if (i >= rows * cols) return;
  int r = i / cols, c = i - r * cols;
  const float* xr = x + (size_t)r * ld_in;
  float xv = xr[c];
  float out = xv;
  if (do_rope) {
    int d = c & 63;                       // position within head
    float f = sign * ts[r] * inv_freq[d >> 1];
    float rot = (d & 1) ? xr[c - 1] : -xr[c + 1];
    out = xv * __cosf(f) + rot * __sinf(f);
  }
  if (trans) {
    int b = r / rpb, n = r - b * rpb;
    y[((size_t)b * cols + c) * rpb + n] = f2bf(out);
  } else {
    y[i] = f2bf(out);
  }
}

// Row softmax with scale, bf16 output. One 256-thread block per row.
__global__ void k_softmax_bf16(const float* __restrict__ s,
                               unsigned short* __restrict__ p,
                               int nk, float scale) {
  __shared__ float red[8];
  long long row = blockIdx.x;
  const float* sr = s + row * (long long)nk;
  unsigned short* pr = p + row * (long long)nk;
  int tid = threadIdx.x, lane = tid & 31, w = tid >> 5;

  float mx = -3.0e38f;
  for (int i = tid; i < nk; i += 256) mx = fmaxf(mx, sr[i] * scale);
#pragma unroll
  for (int m = 16; m >= 1; m >>= 1) mx = fmaxf(mx, __shfl_xor(mx, m, 32));
  if (lane == 0) red[w] = mx;
  __syncthreads();
  mx = red[0];
#pragma unroll
  for (int i = 1; i < 8; ++i) mx = fmaxf(mx, red[i]);
  __syncthreads();

  float sum = 0.f;
  for (int i = tid; i < nk; i += 256) sum += __expf(sr[i] * scale - mx);
#pragma unroll
  for (int m = 16; m >= 1; m >>= 1) sum += __shfl_xor(sum, m, 32);
  if (lane == 0) red[w] = sum;
  __syncthreads();
  float tot = red[0];
#pragma unroll
  for (int i = 1; i < 8; ++i) tot += red[i];
  float inv = 1.f / tot;
  for (int i = tid; i < nk; i += 256)
    pr[i] = f2bf(__expf(sr[i] * scale - mx) * inv);
}

// GEGLU: h (rows x 4096) -> y (rows x 2048) bf16, y = a * gelu(g), exact erf.
__global__ void k_glugelu(const float* __restrict__ h,
                          unsigned short* __restrict__ y, int rows) {
  int i = blockIdx.x * blockDim.x + threadIdx.x;
  if (i >= rows * 2048) return;
  int r = i >> 11, c = i & 2047;
  float a = h[(size_t)r * 4096 + c];
  float g = h[(size_t)r * 4096 + 2048 + c];
  float ge = 0.5f * g * (1.0f + erff(g * 0.7071067811865475f));
  y[i] = f2bf(a * ge);
}

// ---------------------------------------------------------------------------
// WMMA GEMM (NT): C[M,N] f32 = A[M,K]bf16 * B'[N,K]bf16^T  (+bias[N]) (+R)
// Batched: z = b*heads + h ; operand ptr = base + b*bs + h*hs.
// Block: 256 threads = 8 waves; tile 64(M) x 128(N); K-step 64; wave tile 32x32.
// LDS double-buffered, tiles pre-swizzled to per-lane fragments in contiguous k order (k = 16h + s),
// streamed with global_load_async_to_lds_b128 (ASYNCcnt).
// Requires: M % 64 == 0, K % 64 == 0 (N is bounds-guarded / zero-padded).
// ---------------------------------------------------------------------------
__device__ __forceinline__ void fill_tiles(unsigned short* Asb, unsigned short* Bsb,
                                           const unsigned short* Ab, int lda,
                                           const unsigned short* Bb, int ldb,
                                           int m0, int n0, int k0, int N, int tid) {
  // A: 64 rows x 64 K = 8KB: 512 16B-tasks
#pragma unroll
  for (int t2 = 0; t2 < 2; ++t2) {
    int task = tid + t2 * 256;
    int rowl = task >> 3;                 // 0..63
    int oct = task & 7;                   // K octet within 64
    int sk = oct >> 2, o2 = oct & 3;      // 32-chunk, octet within chunk
    int gg = o2 >> 1, h8 = o2 & 1;        // K = sk*32 + gg*16 + h8*8 + 0..7
    int sub = (rowl >> 4) * 2 + sk;
    unsigned short* dst = Asb + sub * 512 + ((rowl & 15) + gg * 16) * 16 + h8 * 8;
    const unsigned short* src = Ab + (long long)(m0 + rowl) * lda + (k0 + oct * 8);
#if USE_ASYNC_COPY
    async_b128(dst, src);
#else
    *(uint4*)dst = *(const uint4*)src;
#endif
  }
  // B': 128 rows x 64 K = 16KB: 512 32B-tasks
#pragma unroll
  for (int t2 = 0; t2 < 2; ++t2) {
    int task = tid + t2 * 256;
    int nl = task >> 2;                   // 0..127
    int ks = task & 3;                    // 16-K slot
    int sk = ks >> 1, gg = ks & 1;
    int gn = n0 + nl;
    int sub = (nl >> 4) * 2 + sk;
    unsigned short* dst = Bsb + sub * 512 + ((nl & 15) + gg * 16) * 16;
    if (gn < N) {
      const unsigned short* src = Bb + (long long)gn * ldb + (k0 + ks * 16);
#if USE_ASYNC_COPY
      async_b128(dst, src);
      async_b128(dst + 8, src + 8);
#else
      *(uint4*)dst = *(const uint4*)src;
      *(uint4*)(dst + 8) = *(const uint4*)(src + 8);
#endif
    } else {
      *(uint4*)dst = make_uint4(0u, 0u, 0u, 0u);
      *(uint4*)(dst + 8) = make_uint4(0u, 0u, 0u, 0u);
    }
  }
}

__device__ __forceinline__ void compute_tiles(const unsigned short* Asb,
                                              const unsigned short* Bsb,
                                              int wm, int wn, int lane,
                                              v8f (&acc)[2][2]) {
#pragma unroll
  for (int sk = 0; sk < 2; ++sk) {
    v16bf a0 = *(const v16bf*)&Asb[((wm * 2 + 0) * 2 + sk) * 512 + lane * 16];
    v16bf a1 = *(const v16bf*)&Asb[((wm * 2 + 1) * 2 + sk) * 512 + lane * 16];
    v16bf b0 = *(const v16bf*)&Bsb[((wn * 2 + 0) * 2 + sk) * 512 + lane * 16];
    v16bf b1 = *(const v16bf*)&Bsb[((wn * 2 + 1) * 2 + sk) * 512 + lane * 16];
    acc[0][0] = __builtin_amdgcn_wmma_f32_16x16x32_bf16(false, a0, false, b0, (short)0, acc[0][0], false, false);
    acc[0][1] = __builtin_amdgcn_wmma_f32_16x16x32_bf16(false, a0, false, b1, (short)0, acc[0][1], false, false);
    acc[1][0] = __builtin_amdgcn_wmma_f32_16x16x32_bf16(false, a1, false, b0, (short)0, acc[1][0], false, false);
    acc[1][1] = __builtin_amdgcn_wmma_f32_16x16x32_bf16(false, a1, false, b1, (short)0, acc[1][1], false, false);
  }
}

__global__ __launch_bounds__(256) void k_gemm_wmma(
    const unsigned short* __restrict__ A, long long a_bs, long long a_hs, int lda,
    const unsigned short* __restrict__ Bm, long long b_bs, long long b_hs, int ldb,
    float* __restrict__ C, long long c_bs, long long c_hs, int ldc,
    const float* __restrict__ bias,
    const float* __restrict__ R, long long r_bs, long long r_hs, int ldr,
    int M, int N, int K, int heads) {
  __shared__ __align__(32) unsigned short As[2][8 * 512];    // 2 x 8KB
  __shared__ __align__(32) unsigned short Bs[2][16 * 512];   // 2 x 16KB

  int z = blockIdx.z;
  int bb = z / heads, hh = z - bb * heads;
  const unsigned short* Ab = A + (long long)bb * a_bs + (long long)hh * a_hs;
  const unsigned short* Bb = Bm + (long long)bb * b_bs + (long long)hh * b_hs;
  float* Cb = C + (long long)bb * c_bs + (long long)hh * c_hs;
  const float* Rb = R ? (R + (long long)bb * r_bs + (long long)hh * r_hs) : (const float*)0;

  int m0 = blockIdx.y * 64;
  int n0 = blockIdx.x * 128;
  int tid = threadIdx.x;
  int lane = tid & 31;
  int wid = tid >> 5;
  int wm = wid >> 2;        // 0..1 -> M sub-block of 32
  int wn = wid & 3;         // 0..3 -> N sub-block of 32

  v8f acc[2][2];
#pragma unroll
  for (int i = 0; i < 2; ++i)
#pragma unroll
    for (int j = 0; j < 2; ++j)
#pragma unroll
      for (int v = 0; v < 8; ++v) acc[i][j][v] = 0.f;

  fill_tiles(As[0], Bs[0], Ab, lda, Bb, ldb, m0, n0, 0, N, tid);
#if USE_ASYNC_COPY
  wait_async0();
#endif
  __syncthreads();

  int iters = K >> 6;
  for (int it = 0; it < iters; ++it) {
    int cur = it & 1;
    if (it + 1 < iters)
      fill_tiles(As[cur ^ 1], Bs[cur ^ 1], Ab, lda, Bb, ldb, m0, n0, (it + 1) << 6, N, tid);
    compute_tiles(As[cur], Bs[cur], wm, wn, lane, acc);
    if (it + 1 < iters) {
#if USE_ASYNC_COPY
      wait_async0();
#endif
      __syncthreads();
    }
  }

  // epilogue: C layout (CDNA5 ISA): VGPR v -> row (lane>=16)*8+v, col lane&15
  int lr = (lane >> 4) * 8;
  int lc = lane & 15;
#pragma unroll
  for (int i = 0; i < 2; ++i) {
#pragma unroll
    for (int j = 0; j < 2; ++j) {
      int cc = n0 + wn * 32 + j * 16 + lc;
      if (cc < N) {
        float bv = bias ? bias[cc] : 0.f;
#pragma unroll
        for (int v = 0; v < 8; ++v) {
          int rr = m0 + wm * 32 + i * 16 + lr + v;
          float val = acc[i][j][v] + bv;
          if (Rb) val += Rb[(long long)rr * ldr + cc];
          Cb[(long long)rr * ldc + cc] = val;
        }
      }
    }
  }
}

// ---------------------------------------------------------------------------
// host orchestration
// ---------------------------------------------------------------------------
extern "C" void kernel_launch(void* const* d_in, const int* in_sizes, int n_in,
                              void* d_out, int out_size, void* d_ws, size_t ws_size,
                              hipStream_t stream) {
  (void)in_sizes; (void)n_in; (void)out_size; (void)ws_size;

  const int Bz = 8, NIN = 4096, NLAT = 512, NOUT = 2048;
  const int ROWS_LAT = Bz * NLAT;   // 4096
  const int ROWS_IN  = Bz * NIN;    // 32768
  const int ROWS_OUT = Bz * NOUT;   // 16384

  // ----- inputs (setup_inputs dict order; params = jax pytree sorted keys)
  const float* inputs   = (const float*)d_in[0];
  const float* latents0 = (const float*)d_in[1];
  const float* outq     = (const float*)d_in[2];
  const float* in_ts    = (const float*)d_in[3];
  const float* lat_ts   = (const float*)d_in[4];
  const float* out_ts   = (const float*)d_in[5];
  const float* inv_freq = (const float*)d_in[6];
  const float* dec_bo   = (const float*)d_in[7];
  const float* dec_lnb  = (const float*)d_in[8];
  const float* dec_lng  = (const float*)d_in[9];
  const float* dec_lncb = (const float*)d_in[10];
  const float* dec_lncg = (const float*)d_in[11];
  const float* dec_wkv  = (const float*)d_in[12];
  const float* dec_wo   = (const float*)d_in[13];
  const float* dec_wq   = (const float*)d_in[14];
  const float* dff_b1   = (const float*)d_in[15];
  const float* dff_b2   = (const float*)d_in[16];
  const float* dff_lnb  = (const float*)d_in[17];
  const float* dff_lng  = (const float*)d_in[18];
  const float* dff_w1   = (const float*)d_in[19];
  const float* dff_w2   = (const float*)d_in[20];
  const float* enc_bo   = (const float*)d_in[21];
  const float* enc_lnb  = (const float*)d_in[22];
  const float* enc_lng  = (const float*)d_in[23];
  const float* enc_lncb = (const float*)d_in[24];
  const float* enc_lncg = (const float*)d_in[25];
  const float* enc_wkv  = (const float*)d_in[26];
  const float* enc_wo   = (const float*)d_in[27];
  const float* enc_wq   = (const float*)d_in[28];
  const float* eff_b1   = (const float*)d_in[29];
  const float* eff_b2   = (const float*)d_in[30];
  const float* eff_lnb  = (const float*)d_in[31];
  const float* eff_lng  = (const float*)d_in[32];
  const float* eff_w1   = (const float*)d_in[33];
  const float* eff_w2   = (const float*)d_in[34];
  const float *lf_b1[2], *lf_b2[2], *lf_lnb[2], *lf_lng[2], *lf_w1[2], *lf_w2[2];
  const float *ls_bo[2], *ls_lnb[2], *ls_lng[2], *ls_wo[2], *ls_wqkv[2];
  for (int L = 0; L < 2; ++L) {
    int base = 35 + L * 11;
    lf_b1[L] = (const float*)d_in[base + 0];
    lf_b2[L] = (const float*)d_in[base + 1];
    lf_lnb[L] = (const float*)d_in[base + 2];
    lf_lng[L] = (const float*)d_in[base + 3];
    lf_w1[L] = (const float*)d_in[base + 4];
    lf_w2[L] = (const float*)d_in[base + 5];
    ls_bo[L] = (const float*)d_in[base + 6];
    ls_lnb[L] = (const float*)d_in[base + 7];
    ls_lng[L] = (const float*)d_in[base + 8];
    ls_wo[L] = (const float*)d_in[base + 9];
    ls_wqkv[L] = (const float*)d_in[base + 10];
  }

  // ----- workspace arena
  size_t off = 0;
  auto alloc = [&](size_t bytes) -> char* {
    char* p = (char*)d_ws + off;
    off += (bytes + 255) & ~(size_t)255;
    return p;
  };
  // f32 (R x C) weight -> bf16 transposed (C x R), so all GEMMs are NT.
  auto cvtT = [&](const float* src, int R, int Ccols) -> unsigned short* {
    unsigned short* dst = (unsigned short*)alloc((size_t)R * Ccols * 2);
    int n = R * Ccols;
    k_cvt_t<<<(n + 255) / 256, 256, 0, stream>>>(src, dst, R, Ccols);
    return dst;
  };

  // bf16 transposed weights
  unsigned short* enc_wq_t  = cvtT(enc_wq, 512, 64);     // (64,512)
  unsigned short* enc_wkv_t = cvtT(enc_wkv, 512, 128);   // (128,512)
  unsigned short* enc_wo_t  = cvtT(enc_wo, 64, 512);     // (512,64)
  unsigned short* dec_wq_t  = cvtT(dec_wq, 512, 64);
  unsigned short* dec_wkv_t = cvtT(dec_wkv, 512, 128);
  unsigned short* dec_wo_t  = cvtT(dec_wo, 64, 512);
  unsigned short* eff_w1_t  = cvtT(eff_w1, 512, 4096);   // (4096,512)
  unsigned short* eff_w2_t  = cvtT(eff_w2, 2048, 512);   // (512,2048)
  unsigned short* dff_w1_t  = cvtT(dff_w1, 512, 4096);
  unsigned short* dff_w2_t  = cvtT(dff_w2, 2048, 512);
  unsigned short *lw_qkv_t[2], *lw_wo_t[2], *lw_w1_t[2], *lw_w2_t[2];
  for (int L = 0; L < 2; ++L) {
    lw_qkv_t[L] = cvtT(ls_wqkv[L], 512, 1536);           // (1536,512)
    lw_wo_t[L]  = cvtT(ls_wo[L], 512, 512);
    lw_w1_t[L]  = cvtT(lf_w1[L], 512, 4096);
    lw_w2_t[L]  = cvtT(lf_w2[L], 2048, 512);
  }

  // activations
  float*          lat   = (float*)alloc((size_t)ROWS_LAT * 512 * 4);
  unsigned short* lnq   = (unsigned short*)alloc((size_t)ROWS_OUT * 512 * 2);
  unsigned short* lnc   = (unsigned short*)alloc((size_t)ROWS_IN * 512 * 2);
  float*          qkvf  = (float*)alloc((size_t)6291456 * 4);
  unsigned short* qbf   = (unsigned short*)alloc((size_t)2097152 * 2);
  unsigned short* kbf   = (unsigned short*)alloc((size_t)2097152 * 2);
  unsigned short* vbfT  = (unsigned short*)alloc((size_t)2097152 * 2);  // transposed V
  float*          S     = (float*)alloc((size_t)16777216 * 4);
  unsigned short* Pb    = (unsigned short*)alloc((size_t)16777216 * 2);
  float*          attnf = (float*)alloc((size_t)2097152 * 4);
  unsigned short* attnb = (unsigned short*)alloc((size_t)2097152 * 2);
  float*          hid   = (float*)alloc((size_t)8388608 * 4);
  unsigned short* hbf   = (unsigned short*)alloc((size_t)4194304 * 2);

  // launch wrappers -------------------------------------------------------
  auto GEMM = [&](const unsigned short* Am, long long abs_, long long ahs, int lda,
                  const unsigned short* Bm, long long bbs, long long bhs, int ldb,
                  float* Cm, long long cbs, long long chs, int ldc,
                  const float* bias, const float* R, long long rbs, long long rhs, int ldr,
                  int M, int N, int K, int batch, int heads) {
    dim3 g((unsigned)((N + 127) / 128), (unsigned)(M / 64), (unsigned)(batch * heads));
    k_gemm_wmma<<<g, 256, 0, stream>>>(Am, abs_, ahs, lda, Bm, bbs, bhs, ldb,
                                       Cm, cbs, chs, ldc, bias, R, rbs, rhs, ldr,
                                       M, N, K, heads);
  };
  auto LN = [&](const float* x, const float* g, const float* b, unsigned short* y, int rows) {
    k_layernorm_bf16<<<rows / 4, 128, 0, stream>>>(x, g, b, y, rows);
  };
  auto ROPE = [&](const float* x, int ld, unsigned short* y, const float* ts,
                  float sign, int rows, int cols, int dorope, int trans, int rpb) {
    int n = rows * cols;
    k_rope_bf16<<<(n + 255) / 256, 256, 0, stream>>>(x, ld, y, ts, inv_freq, sign,
                                                     rows, cols, dorope, trans, rpb);
  };
  auto FFN = [&](float* xres, int rows, const float* g, const float* b,
                 const unsigned short* w1t, const float* b1,
                 const unsigned short* w2t, const float* b2) {
    LN(xres, g, b, lnq, rows);
    for (int r0 = 0; r0 < rows; r0 += 2048) {
      GEMM(lnq + (size_t)r0 * 512, 0, 0, 512, w1t, 0, 0, 512,
           hid, 0, 0, 4096, b1, nullptr, 0, 0, 0, 2048, 4096, 512, 1, 1);
      k_glugelu<<<(2048 * 2048 + 255) / 256, 256, 0, stream>>>(hid, hbf, 2048);
      GEMM(hbf, 0, 0, 2048, w2t, 0, 0, 2048, xres + (size_t)r0 * 512, 0, 0, 512,
           b2, xres + (size_t)r0 * 512, 0, 0, 512, 2048, 512, 2048, 1, 1);
    }
  };

  // residual streams
  k_copy_f32<<<(ROWS_LAT * 512 + 255) / 256, 256, 0, stream>>>(latents0, lat, ROWS_LAT * 512);
  float* outS = (float*)d_out;
  k_copy_f32<<<(ROWS_OUT * 512 + 255) / 256, 256, 0, stream>>>(outq, outS, ROWS_OUT * 512);

  // ================= encoder cross-attn (heads=1, rotate_value=True) =======
  LN(lat, enc_lng, enc_lnb, lnq, ROWS_LAT);
  LN(inputs, enc_lncg, enc_lncb, lnc, ROWS_IN);
  float* Qf = qkvf;
  float* KVf = qkvf + (size_t)ROWS_LAT * 64;
  GEMM(lnq, 0, 0, 512, enc_wq_t, 0, 0, 512, Qf, 0, 0, 64,
       nullptr, nullptr, 0, 0, 0, ROWS_LAT, 64, 512, 1, 1);
  GEMM(lnc, 0, 0, 512, enc_wkv_t, 0, 0, 512, KVf, 0, 0, 128,
       nullptr, nullptr, 0, 0, 0, ROWS_IN, 128, 512, 1, 1);
  ROPE(Qf, 64, qbf, lat_ts, 1.f, ROWS_LAT, 64, 1, 0, NLAT);
  ROPE(KVf, 128, kbf, in_ts, 1.f, ROWS_IN, 64, 1, 0, NIN);
  ROPE(KVf + 64, 128, vbfT, in_ts, 1.f, ROWS_IN, 64, 1, 1, NIN);   // V^T
  GEMM(qbf, (long long)NLAT * 64, 0, 64, kbf, (long long)NIN * 64, 0, 64,
       S, (long long)NLAT * NIN, 0, NIN, nullptr, nullptr, 0, 0, 0,
       NLAT, NIN, 64, Bz, 1);
  k_softmax_bf16<<<Bz * NLAT, 256, 0, stream>>>(S, Pb, NIN, 0.125f);
  GEMM(Pb, (long long)NLAT * NIN, 0, NIN, vbfT, (long long)64 * NIN, 0, NIN,
       attnf, (long long)NLAT * 64, 0, 64, nullptr, nullptr, 0, 0, 0,
       NLAT, 64, NIN, Bz, 1);
  ROPE(attnf, 64, attnb, lat_ts, -1.f, ROWS_LAT, 64, 1, 0, NLAT);  // un-rotate
  GEMM(attnb, 0, 0, 64, enc_wo_t, 0, 0, 64, lat, 0, 0, 512,
       enc_bo, lat, 0, 0, 512, ROWS_LAT, 512, 64, 1, 1);
  FFN(lat, ROWS_LAT, eff_lng, eff_lnb, eff_w1_t, eff_b1, eff_w2_t, eff_b2);

  // ================= self-attn layers (heads=8, rotate_value=True) =========
  for (int L = 0; L < 2; ++L) {
    LN(lat, ls_lng[L], ls_lnb[L], lnq, ROWS_LAT);
    GEMM(lnq, 0, 0, 512, lw_qkv_t[L], 0, 0, 512, qkvf, 0, 0, 1536,
         nullptr, nullptr, 0, 0, 0, ROWS_LAT, 1536, 512, 1, 1);
    ROPE(qkvf, 1536, qbf, lat_ts, 1.f, ROWS_LAT, 512, 1, 0, NLAT);
    ROPE(qkvf + 512, 1536, kbf, lat_ts, 1.f, ROWS_LAT, 512, 1, 0, NLAT);
    ROPE(qkvf + 1024, 1536, vbfT, lat_ts, 1.f, ROWS_LAT, 512, 1, 1, NLAT);  // V^T
    GEMM(qbf, (long long)NLAT * 512, 64, 512, kbf, (long long)NLAT * 512, 64, 512,
         S, (long long)8 * NLAT * NLAT, (long long)NLAT * NLAT, NLAT,
         nullptr, nullptr, 0, 0, 0, NLAT, NLAT, 64, Bz, 8);
    k_softmax_bf16<<<Bz * 8 * NLAT, 256, 0, stream>>>(S, Pb, NLAT, 0.125f);
    GEMM(Pb, (long long)8 * NLAT * NLAT, (long long)NLAT * NLAT, NLAT,
         vbfT, (long long)512 * NLAT, (long long)64 * NLAT, NLAT,
         attnf, (long long)NLAT * 512, 64, 512,
         nullptr, nullptr, 0, 0, 0, NLAT, 64, NLAT, Bz, 8);
    ROPE(attnf, 512, attnb, lat_ts, -1.f, ROWS_LAT, 512, 1, 0, NLAT);
    GEMM(attnb, 0, 0, 512, lw_wo_t[L], 0, 0, 512, lat, 0, 0, 512,
         ls_bo[L], lat, 0, 0, 512, ROWS_LAT, 512, 512, 1, 1);
    FFN(lat, ROWS_LAT, lf_lng[L], lf_lnb[L], lw_w1_t[L], lf_b1[L], lw_w2_t[L], lf_b2[L]);
  }

  // ================= decoder cross-attn (heads=1, rotate_value=False) ======
  LN(outS, dec_lng, dec_lnb, lnq, ROWS_OUT);
  LN(lat, dec_lncg, dec_lncb, lnc, ROWS_LAT);
  Qf = qkvf;
  KVf = qkvf + (size_t)ROWS_OUT * 64;
  GEMM(lnq, 0, 0, 512, dec_wq_t, 0, 0, 512, Qf, 0, 0, 64,
       nullptr, nullptr, 0, 0, 0, ROWS_OUT, 64, 512, 1, 1);
  GEMM(lnc, 0, 0, 512, dec_wkv_t, 0, 0, 512, KVf, 0, 0, 128,
       nullptr, nullptr, 0, 0, 0, ROWS_LAT, 128, 512, 1, 1);
  ROPE(Qf, 64, qbf, out_ts, 1.f, ROWS_OUT, 64, 1, 0, NOUT);
  ROPE(KVf, 128, kbf, lat_ts, 1.f, ROWS_LAT, 64, 1, 0, NLAT);
  ROPE(KVf + 64, 128, vbfT, lat_ts, 0.f, ROWS_LAT, 64, 0, 1, NLAT);  // V^T, no rope
  GEMM(qbf, (long long)NOUT * 64, 0, 64, kbf, (long long)NLAT * 64, 0, 64,
       S, (long long)NOUT * NLAT, 0, NLAT, nullptr, nullptr, 0, 0, 0,
       NOUT, NLAT, 64, Bz, 1);
  k_softmax_bf16<<<Bz * NOUT, 256, 0, stream>>>(S, Pb, NLAT, 0.125f);
  GEMM(Pb, (long long)NOUT * NLAT, 0, NLAT, vbfT, (long long)64 * NLAT, 0, NLAT,
       attnf, (long long)NOUT * 64, 0, 64, nullptr, nullptr, 0, 0, 0,
       NOUT, 64, NLAT, Bz, 1);
  ROPE(attnf, 64, attnb, out_ts, 0.f, ROWS_OUT, 64, 0, 0, NOUT);    // convert only
  GEMM(attnb, 0, 0, 64, dec_wo_t, 0, 0, 64, outS, 0, 0, 512,
       dec_bo, outS, 0, 0, 512, ROWS_OUT, 512, 64, 1, 1);
  FFN(outS, ROWS_OUT, dff_lng, dff_lnb, dff_w1_t, dff_b1, dff_w2_t, dff_b2);
}
